// FusedSparseMKVAttention_18734647345481
// MI455X (gfx1250) — matrix operand order repaired and checked by reading, not compiled, not run on hardware
//
#include <hip/hip_runtime.h>

typedef __bf16 bf16;
typedef __attribute__((ext_vector_type(16))) __bf16 v16bf;
typedef __attribute__((ext_vector_type(8)))  __bf16 v8bf;
typedef __attribute__((ext_vector_type(4)))  __bf16 v4bf;
typedef __attribute__((ext_vector_type(8)))  float  v8f;
typedef __attribute__((ext_vector_type(4)))  float  v4f;
typedef __attribute__((ext_vector_type(4)))  int    v4i;

#if defined(__AMDGCN__) && __has_builtin(__builtin_amdgcn_global_load_async_to_lds_b128) && \
    __has_builtin(__builtin_amdgcn_s_wait_asynccnt)
#define USE_ASYNC_LDS 1
#else
#define USE_ASYNC_LDS 0
#endif

union FragBF { v16bf v; v8bf h[2]; };

__device__ __forceinline__ v8f wmma_bf16(const FragBF& a, const FragBF& b, v8f c) {
  // D = A(16x32 bf16) * B(32x16 bf16) + C(16x16 f32)
  return __builtin_amdgcn_wmma_f32_16x16x32_bf16(
      /*neg_a=*/false, a.v, /*neg_b=*/false, b.v,
      /*c_mod=*/(short)0, c, /*reuse_a=*/false, /*reuse_b=*/false);
}

__device__ __forceinline__ v8f zero8() {
  v8f z;
#pragma unroll
  for (int e = 0; e < 8; ++e) z[e] = 0.f;
  return z;
}

#if USE_ASYNC_LDS
typedef __attribute__((address_space(1))) v4i* global_v4i_ptr;  // "__device__" in diags
typedef __attribute__((address_space(3))) v4i* lds_v4i_ptr;     // "__shared__" in diags

__device__ __forceinline__ void async_b128(const bf16* g, bf16* lds) {
  __builtin_amdgcn_global_load_async_to_lds_b128(
      (global_v4i_ptr)g, (lds_v4i_ptr)lds, /*offset=*/0, /*cpol=*/0);
}
#endif

// ---------------------------------------------------------------------------
// f32 -> bf16 convert (vectorized by 4)
// ---------------------------------------------------------------------------
__global__ __launch_bounds__(256) void cvt_f32_to_bf16(const float* __restrict__ in,
                                                       bf16* __restrict__ out, int n4) {
  int i = blockIdx.x * 256 + threadIdx.x;
  if (i < n4) {
    v4f f = ((const v4f*)in)[i];
    v4bf o;
    o.x = (bf16)f.x; o.y = (bf16)f.y; o.z = (bf16)f.z; o.w = (bf16)f.w;
    ((v4bf*)out)[i] = o;
  }
}

// V_mem (B,64,2048) f32  ->  VbT (B,2048,64) bf16   (so att@V becomes A@B^T)
__global__ __launch_bounds__(256) void transpose_cvt_v(const float* __restrict__ V,
                                                       bf16* __restrict__ VbT) {
  int idx = blockIdx.x * 256 + threadIdx.x;     // exactly 4*64*2048 threads
  int d = idx & 2047;
  int n = (idx >> 11) & 63;
  int b = idx >> 17;
  VbT[((size_t)b * 2048 + d) * 64 + n] = (bf16)V[idx];
}

// ---------------------------------------------------------------------------
// Stage one 128x32 A tile + 128x32 B tile into LDS (bf16, 40-elem padded rows).
// bf16 operands go through GLOBAL_LOAD_ASYNC_TO_LDS_B128 when available;
// f32 A is loaded to VGPRs, converted, and ds-stored.
// ---------------------------------------------------------------------------
template <typename TA>
__device__ __forceinline__ void stage_tile(const TA* __restrict__ Ab, int lda,
                                           const bf16* __restrict__ Bb, int ldb,
                                           int m0, int n0, int k0, int srow, int cseg,
                                           bf16 (*As)[40], bf16 (*Bs)[40]) {
  {  // A tile
    const TA* src = Ab + (size_t)(m0 + srow) * lda + k0 + cseg;
    if constexpr (sizeof(TA) == 4) {
      v4f f0 = ((const v4f*)src)[0];
      v4f f1 = ((const v4f*)src)[1];
      v4f f2 = ((const v4f*)src)[2];
      v4f f3 = ((const v4f*)src)[3];
      v8bf o0, o1;
#pragma unroll
      for (int e = 0; e < 4; ++e) {
        o0[e] = (bf16)f0[e]; o0[e + 4] = (bf16)f1[e];
        o1[e] = (bf16)f2[e]; o1[e + 4] = (bf16)f3[e];
      }
      *(v8bf*)&As[srow][cseg]     = o0;
      *(v8bf*)&As[srow][cseg + 8] = o1;
    } else {
#if USE_ASYNC_LDS
      async_b128((const bf16*)src,     &As[srow][cseg]);
      async_b128((const bf16*)src + 8, &As[srow][cseg + 8]);
#else
      *(v8bf*)&As[srow][cseg]     = ((const v8bf*)src)[0];
      *(v8bf*)&As[srow][cseg + 8] = ((const v8bf*)src)[1];
#endif
    }
  }
  {  // B tile (rows are N, cols are K)
    const bf16* src = Bb + (size_t)(n0 + srow) * ldb + k0 + cseg;
#if USE_ASYNC_LDS
    async_b128(src,     &Bs[srow][cseg]);
    async_b128(src + 8, &Bs[srow][cseg + 8]);
#else
    *(v8bf*)&Bs[srow][cseg]     = ((const v8bf*)src)[0];
    *(v8bf*)&Bs[srow][cseg + 8] = ((const v8bf*)src)[1];
#endif
  }
}

// ---------------------------------------------------------------------------
// Tiled WMMA GEMM:  C[M x N] = A[M x K] * Bm[N x K]^T   (both K-major)
// BM=BN=128, BK=32, 256 threads = 8 waves in 2(M) x 4(N); wave tile 64x32.
// Double-buffered LDS; tile k+1 streamed via async-to-LDS while tile k runs
// on the WMMA pipes; s_wait_asynccnt leaves the prefetch in flight.
// ---------------------------------------------------------------------------
template <typename TA, typename TOUT>
__global__ __launch_bounds__(256) void gemm_wmma_bf16(
    const TA* __restrict__ A, int lda, long long sA,
    const bf16* __restrict__ Bm, int ldb, long long sB,
    TOUT* __restrict__ C, int ldc, long long sC, int K) {
  __shared__ bf16 As[2][128][40];   // +8 pad keeps 16B alignment, dodges conflicts
  __shared__ bf16 Bs[2][128][40];

  const int m0 = blockIdx.y * 128;
  const int n0 = blockIdx.x * 128;
  const int zb = blockIdx.z;
  const TA*   Ab = A  + (size_t)zb * sA;
  const bf16* Bb = Bm + (size_t)zb * sB;
  TOUT*       Cb = C  + (size_t)zb * sC;

  const int tid   = threadIdx.x;
  const int srow  = tid >> 1;           // staging row 0..127
  const int cseg  = (tid & 1) * 16;     // staging col base {0,16}

  const int lane  = tid & 31;
  const int wave  = tid >> 5;
  const int waveM = wave >> 2;          // 0..1
  const int waveN = wave & 3;           // 0..3
  const int lrow  = lane & 15;
  const int half  = lane >> 4;

  // async loads issued per thread per staged tile (immediate for s_wait_asynccnt)
  constexpr short NASYNC = (sizeof(TA) == 4) ? 2 : 4;

  v8f acc[4][2];
#pragma unroll
  for (int i = 0; i < 4; ++i)
#pragma unroll
    for (int j = 0; j < 2; ++j) acc[i][j] = zero8();

  const int KT = K / 32;
  stage_tile<TA>(Ab, lda, Bb, ldb, m0, n0, 0, srow, cseg, As[0], Bs[0]);

  for (int kt = 0; kt < KT; ++kt) {
    const int cur = kt & 1;
    if (kt + 1 < KT) {
      // prefetch next tile into the other buffer, then wait only for tile kt
      stage_tile<TA>(Ab, lda, Bb, ldb, m0, n0, (kt + 1) * 32, srow, cseg,
                     As[cur ^ 1], Bs[cur ^ 1]);
#if USE_ASYNC_LDS
      __builtin_amdgcn_s_wait_asynccnt(NASYNC);
#endif
    } else {
#if USE_ASYNC_LDS
      __builtin_amdgcn_s_wait_asynccnt(0);
#endif
    }
    __syncthreads();

    FragBF af[4];
#pragma unroll
    for (int i = 0; i < 4; ++i) {
      const int ar = waveM * 64 + i * 16 + lrow;
      af[i].h[0] = *(const v8bf*)&As[cur][ar][half * 8];        // K=k0+half*8+0..7
      af[i].h[1] = *(const v8bf*)&As[cur][ar][16 + half * 8];   // K=k0+16+half*8+0..7
    }
    FragBF bfr[2];
#pragma unroll
    for (int j = 0; j < 2; ++j) {
      const int br = waveN * 32 + j * 16 + lrow;                // column n
      bfr[j].h[0] = *(const v8bf*)&Bs[cur][br][half * 8];       // K=half*8+0..7
      bfr[j].h[1] = *(const v8bf*)&Bs[cur][br][16 + half * 8];  // K=16+half*8+0..7
    }
#pragma unroll
    for (int i = 0; i < 4; ++i)
#pragma unroll
      for (int j = 0; j < 2; ++j) acc[i][j] = wmma_bf16(af[i], bfr[j], acc[i][j]);

    __syncthreads();  // all waves done reading buf `cur` before it is re-staged
  }

  // epilogue: C/D layout -> row = base + half*8 + r, col = base + lane%16
#pragma unroll
  for (int i = 0; i < 4; ++i)
#pragma unroll
    for (int j = 0; j < 2; ++j) {
      const int row = m0 + waveM * 64 + i * 16 + half * 8;
      const int col = n0 + waveN * 32 + j * 16 + lrow;
      TOUT* dst = Cb + (size_t)row * ldc + col;
#pragma unroll
      for (int r = 0; r < 8; ++r) dst[(size_t)r * ldc] = (TOUT)acc[i][j][r];
    }
}

// ---------------------------------------------------------------------------
// Fused scores + top-k(54 of 64) threshold + softmax  ->  att (B,S,64) bf16
// One wave = one 16-row tile. 4 WMMA accumulators cover N=64, K loop over 2048.
// 54th largest of 64 == 11th smallest -> 11-pass min selection (index tiebreak).
// ---------------------------------------------------------------------------
__global__ __launch_bounds__(256) void scores_topk_softmax(
    const bf16* __restrict__ Qb, const bf16* __restrict__ Kb, bf16* __restrict__ att,
    const float* __restrict__ beta_q, const float* __restrict__ beta_k,
    const float* __restrict__ gate, const float* __restrict__ log_tau) {
  __shared__ float sc[8][16][64];
  __shared__ float rmax[8][16], rthr[8][16], rinv[8][16];

  const int lane = threadIdx.x & 31;
  const int wave = threadIdx.x >> 5;
  const int lrow = lane & 15;
  const int half = lane >> 4;

  const int tile = blockIdx.x * 8 + wave;   // 0..1023 = B * S/16
  const int b    = tile >> 8;
  const int m0   = (tile & 255) << 4;

  const float g     = *gate;
  const float alpha = 1.f / (1.f + __expf(-g));
  const float tau   = __expf(*log_tau);
  const float coef  = (alpha * (*beta_q) + (1.f - alpha) * (*beta_k)) * tau *
                      0.02209708691207961f;  // 2048^-0.5

  v8f acc[4];
#pragma unroll
  for (int j = 0; j < 4; ++j) acc[j] = zero8();

  const bf16* qrow  = Qb + (size_t)(b * 4096 + m0 + lrow) * 2048;
  const bf16* kbase = Kb + (size_t)b * 64 * 2048;
  for (int k0 = 0; k0 < 2048; k0 += 32) {
    FragBF a;
    a.h[0] = *(const v8bf*)(qrow + k0 + half * 8);
    a.h[1] = *(const v8bf*)(qrow + k0 + 16 + half * 8);
#pragma unroll
    for (int j = 0; j < 4; ++j) {
      const bf16* kr = kbase + (size_t)(j * 16 + lrow) * 2048 + k0 + half * 8;
      FragBF bfr;
      bfr.h[0] = *(const v8bf*)kr;
      bfr.h[1] = *(const v8bf*)(kr + 16);
      acc[j] = wmma_bf16(a, bfr, acc[j]);
    }
  }

#pragma unroll
  for (int j = 0; j < 4; ++j)
#pragma unroll
    for (int r = 0; r < 8; ++r)
      sc[wave][half * 8 + r][j * 16 + lrow] = coef * acc[j][r];
  __syncthreads();

  if (lane < 16) {
    const float* row = sc[wave][lane];
    float mx = -3.4e38f;
    for (int c = 0; c < 64; ++c) mx = fmaxf(mx, row[c]);
    float lastv = -3.4e38f; int lasti = -1;
    for (int it = 0; it < 11; ++it) {        // 11th smallest = 54th largest
      float bv = 3.4e38f; int bi = 64;
      for (int c = 0; c < 64; ++c) {
        float v = row[c];
        if ((v > lastv) || (v == lastv && c > lasti)) {
          if (v < bv) { bv = v; bi = c; }
        }
      }
      lastv = bv; lasti = bi;
    }
    const float thr = lastv;
    float s = 0.f;
    for (int c = 0; c < 64; ++c) {
      float v = row[c];
      if (v >= thr) s += __expf(v - mx);
    }
    rmax[wave][lane] = mx;
    rthr[wave][lane] = thr;
    rinv[wave][lane] = 1.f / s;
  }
  __syncthreads();

  for (int r = 0; r < 16; ++r) {
    const float mx = rmax[wave][r], thr = rthr[wave][r], inv = rinv[wave][r];
#pragma unroll
    for (int cc = 0; cc < 2; ++cc) {
      const int c = lane + cc * 32;
      const float v = sc[wave][r][c];
      const float p = (v >= thr) ? __expf(v - mx) * inv : 0.f;
      att[(size_t)(b * 4096 + m0 + r) * 64 + c] = (bf16)p;
    }
  }
}

// ---------------------------------------------------------------------------
// Launch: inputs = x, W_qkv, W_o, beta_q, beta_k, gate, log_tau, K_mem, V_mem, clip_k
// ---------------------------------------------------------------------------
extern "C" void kernel_launch(void* const* d_in, const int* in_sizes, int n_in,
                              void* d_out, int out_size, void* d_ws, size_t ws_size,
                              hipStream_t stream) {
  const float* x       = (const float*)d_in[0];
  const float* W_qkv   = (const float*)d_in[1];
  const float* W_o     = (const float*)d_in[2];
  const float* beta_q  = (const float*)d_in[3];
  const float* beta_k  = (const float*)d_in[4];
  const float* gate    = (const float*)d_in[5];
  const float* log_tau = (const float*)d_in[6];
  const float* K_mem   = (const float*)d_in[7];
  const float* V_mem   = (const float*)d_in[8];
  // d_in[9] = clip_k == 54 (hardcoded: 64-54+1 = 11th smallest)

  char* ws = (char*)d_ws;
  const size_t MB = 1024 * 1024;
  bf16* Wq  = (bf16*)(ws);             //  8 MB: W_qkv rows 0..2047 (Q block) bf16
  bf16* Wo  = (bf16*)(ws + 8 * MB);    //  8 MB
  bf16* Kb  = (bf16*)(ws + 16 * MB);   //  1 MB: K_mem bf16
  bf16* VbT = (bf16*)(ws + 17 * MB);   //  1 MB: V_mem^T bf16 (B,2048,64)
  bf16* Qb  = (bf16*)(ws + 18 * MB);   // 64 MB: Q bf16 (16384,2048)
  bf16* att = (bf16*)(ws + 82 * MB);   //  2 MB: attention (B,4096,64)
  bf16* O1  = (bf16*)(ws + 84 * MB);   // 64 MB: att@V bf16
  float* out = (float*)d_out;

  cvt_f32_to_bf16<<<4096, 256, 0, stream>>>(W_qkv, Wq, 1048576);  // 2048*2048/4
  cvt_f32_to_bf16<<<4096, 256, 0, stream>>>(W_o,   Wo, 1048576);
  cvt_f32_to_bf16<<<512,  256, 0, stream>>>(K_mem, Kb, 131072);   // 4*64*2048/4
  transpose_cvt_v<<<2048, 256, 0, stream>>>(V_mem, VbT);

  // GEMM1: Q = x @ Wq^T   (M=16384, N=2048, K=2048), f32 A converted in staging
  gemm_wmma_bf16<float, bf16><<<dim3(16, 128, 1), 256, 0, stream>>>(
      x, 2048, 0LL, Wq, 2048, 0LL, Qb, 2048, 0LL, 2048);

  // scores + top-k + softmax -> att
  scores_topk_softmax<<<128, 256, 0, stream>>>(Qb, Kb, att, beta_q, beta_k, gate, log_tau);

  // GEMM2 (batched): O1_b = att_b @ VbT_b^T  (M=4096, N=2048, K=64)
  gemm_wmma_bf16<bf16, bf16><<<dim3(16, 32, 4), 256, 0, stream>>>(
      att, 64, 4096LL * 64, VbT, 64, 2048LL * 64, O1, 2048, 4096LL * 2048, 64);

  // GEMM3: out = O1 @ Wo^T  (M=16384, N=2048, K=2048), f32 output
  gemm_wmma_bf16<bf16, float><<<dim3(16, 128, 1), 256, 0, stream>>>(
      O1, 2048, 0LL, Wo, 2048, 0LL, out, 2048, 0LL, 2048);
}
